// RadarPillarAttentionBlock_50139448213692
// MI455X (gfx1250) — compile-verified, matrix operand order checked by reading, not run on hardware
//
#include <hip/hip_runtime.h>
#include <hip/hip_bf16.h>
#include <stdint.h>

// ---------------------------------------------------------------------------
// Problem constants (match reference)
// ---------------------------------------------------------------------------
#define kB   2
#define kC   256
#define kHW  (160 * 160)
#define kN   2048
#define kNH  8
#define kDH  32

typedef __bf16 bf16_t;
typedef unsigned int u32;
typedef __attribute__((ext_vector_type(16))) __bf16 v16bf;
typedef __attribute__((ext_vector_type(8)))  __bf16 v8bf;
typedef __attribute__((ext_vector_type(8)))  float  v8f;
typedef __attribute__((ext_vector_type(4)))  u32    v4u;
typedef __attribute__((ext_vector_type(8)))  int    v8i;
typedef __attribute__((ext_vector_type(4)))  int    v4i;

// ---------------------------------------------------------------------------
// Tensor Data Mover: 2D tile load Global -> LDS (CDNA5 TDM, ISA ch.8).
// D# group0: count=1, lds_addr, 57-bit global addr, type=2 ("image").
// D# group1: data_size=2B, tensor dims == tile dims (no OOB), row stride.
// Remaining descriptor slots zero (tensor rank <= 2).  Tracked by TENSORcnt.
// This toolchain's builtin is the 6-arg clang-23 form:
//   (uint32x4, int32x8, int32x4, int32x4, int32x8, i32 cpol)
// ---------------------------------------------------------------------------
__device__ __forceinline__ u32 lds_off_of(const void* p) {
  // Generic LDS pointers carry the workgroup LDS offset in the low 32 bits.
  return (u32)(uintptr_t)p;
}

__device__ __forceinline__ void tdm_load_2d(u32 lds_off, const void* gptr,
                                            u32 tile_w, u32 tile_h, u32 stride0) {
  const unsigned long long ga = (unsigned long long)(uintptr_t)gptr;
  v4u g0;
  g0[0] = 1u;                                              // count=1 (valid)
  g0[1] = lds_off;                                         // lds_addr
  g0[2] = (u32)ga;                                         // global_addr[31:0]
  g0[3] = (u32)((ga >> 32) & 0x01FFFFFFu) | (2u << 30);    // addr[56:32] | type=2
  v8i g1;
  g1[0] = (int)(1u << 16);                                 // data_size=1 -> 2 bytes
  g1[1] = (int)((tile_w & 0xFFFFu) << 16);                 // tensor_dim0[15:0]
  g1[2] = (int)(((tile_w >> 16) & 0xFFFFu) |               // tensor_dim0[31:16]
                ((tile_h & 0xFFFFu) << 16));               // tensor_dim1[15:0]
  g1[3] = (int)(((tile_h >> 16) & 0xFFFFu) |               // tensor_dim1[31:16]
                ((tile_w & 0xFFFFu) << 16));               // tile_dim0
  g1[4] = (int)(tile_h & 0xFFFFu);                         // tile_dim1 (tile_dim2=0)
  g1[5] = (int)stride0;                                    // tensor_dim0_stride[31:0]
  g1[6] = 0;                                               // stride0[47:32] | dim1_stride lo
  g1[7] = 0;
  const v4i z4 = {0, 0, 0, 0};
  const v8i z8 = {0, 0, 0, 0, 0, 0, 0, 0};
  __builtin_amdgcn_tensor_load_to_lds(g0, g1, z4, z4, z8, 0);
}

template <int N>
__device__ __forceinline__ void wait_tensorcnt() {
  __builtin_amdgcn_s_wait_tensorcnt((unsigned short)N);
}

// ---------------------------------------------------------------------------
// Fragment loader for V_WMMA_F32_16X16X32_BF16 (wave32), ISA 7.12.2 layout:
// lanes 0-15: row=lane, K chunks [0..7],[16..23]; lanes 16-31: same rows,
// K chunks [8..15],[24..31].  Works for A (row-major MxK) and B (row-major
// transposed NxK).  Two contiguous 128-bit loads per lane.
// ---------------------------------------------------------------------------
__device__ __forceinline__ v16bf load_frag16x32(const bf16_t* __restrict__ base,
                                                int ld, int lane) {
  const int r  = lane & 15;
  const int hi = (lane >> 4) & 1;
  const bf16_t* p = base + (size_t)r * ld + hi * 8;
  union { v16bf v; v8bf h[2]; } u;
  u.h[0] = *(const v8bf*)(p);
  u.h[1] = *(const v8bf*)(p + 16);
  return u.v;
}

__device__ __forceinline__ v8f wmma_bf16(v16bf a, v16bf b, v8f c) {
  return __builtin_amdgcn_wmma_f32_16x16x32_bf16(
      false, a, false, b, (short)0, c, false, false);
}

// ---------------------------------------------------------------------------
// Kernel 1: transpose+convert the four 256x256 f32 weights into bf16 (N x K)
// ---------------------------------------------------------------------------
__global__ void rpab_prep_weights(const float* __restrict__ Wq,
                                  const float* __restrict__ Wk,
                                  const float* __restrict__ Wv,
                                  const float* __restrict__ Wo,
                                  bf16_t* __restrict__ WT) {
  const int w = blockIdx.y;
  const float* W = (w == 0) ? Wq : (w == 1) ? Wk : (w == 2) ? Wv : Wo;
  const int e = blockIdx.x * blockDim.x + threadIdx.x;
  const int n = e >> 8;
  const int k = e & 255;
  WT[(size_t)w * kC * kC + (size_t)n * kC + k] = (bf16_t)W[(size_t)k * kC + n];
}

// ---------------------------------------------------------------------------
// Kernel 2: compact non-empty pillars; gather tokens (f32 + bf16 copies).
// ---------------------------------------------------------------------------
__global__ void rpab_gather(const float* __restrict__ x,
                            float* __restrict__ tokF,
                            bf16_t* __restrict__ tokB,
                            int* __restrict__ idx,
                            int* __restrict__ cnt) {
  const int b   = blockIdx.y;
  const int pos = blockIdx.x * blockDim.x + threadIdx.x;
  const float* xb = x + (size_t)b * kC * kHW;
  bool nz = false;
  for (int c = 0; c < kC; ++c) nz = nz || (xb[(size_t)c * kHW + pos] != 0.0f);
  if (!nz) return;
  const int n = atomicAdd(&cnt[b], 1);
  if (n >= kN) return;
  idx[b * kN + n] = pos;
  float*  tf = tokF + ((size_t)b * kN + n) * kC;
  bf16_t* tb = tokB + ((size_t)b * kN + n) * kC;
  for (int c = 0; c < kC; ++c) {
    const float v = xb[(size_t)c * kHW + pos];
    tf[c] = v;
    tb[c] = (bf16_t)v;
  }
}

// ---------------------------------------------------------------------------
// GEMM core: block of 8 waves computes 64 rows x 256 cols of A(2048x256) x
// W(256x256).  The 256x32 weight K-slice is DMA'd into LDS by the TDM
// (double-buffered; wave 0 issues, TENSORcnt pipelines tile i+1 under the
// WMMA chain of tile i).  B fragments then come from LDS (ds_load_b128).
// Wave w: rows (blk*4 + w>>1)*16, cols (w&1)*128.
// ---------------------------------------------------------------------------
__device__ __forceinline__ void gemm_tdm(const bf16_t* __restrict__ A,
                                         const bf16_t* __restrict__ Bt,
                                         bf16_t* __restrict__ Wtile0,
                                         bf16_t* __restrict__ Wtile1,
                                         int lane, int wave, int cbase,
                                         v8f acc[8]) {
  const v8f zero = {0.f, 0.f, 0.f, 0.f, 0.f, 0.f, 0.f, 0.f};
#pragma unroll
  for (int t = 0; t < 8; ++t) acc[t] = zero;

  bf16_t* bufs[2] = {Wtile0, Wtile1};
  if (wave == 0)
    tdm_load_2d(lds_off_of(Wtile0), Bt, 32, kC, kC);

  for (int ks = 0; ks < 8; ++ks) {
    const int cur = ks & 1;
    if (wave == 0) {
      if (ks < 7) {
        tdm_load_2d(lds_off_of(bufs[cur ^ 1]), Bt + (ks + 1) * 32, 32, kC, kC);
        wait_tensorcnt<1>();                 // tile ks complete; ks+1 in flight
      } else {
        wait_tensorcnt<0>();
      }
    }
    __syncthreads();                         // tile ks visible in LDS
    v16bf af = load_frag16x32(A + ks * 32, kC, lane);
    v16bf bfr[8];
#pragma unroll
    for (int t = 0; t < 8; ++t)
      bfr[t] = load_frag16x32(bufs[cur] + (size_t)(cbase + t * 16) * 32, 32, lane);
    __syncthreads();                         // buffer free for tile ks+2
#pragma unroll
    for (int t = 0; t < 8; ++t) acc[t] = wmma_bf16(af, bfr[t], acc[t]);
  }
}

// ---------------------------------------------------------------------------
// Kernel 3: Q/K/V projections.  Q pre-scaled by 1/sqrt(dh); V transposed.
// ---------------------------------------------------------------------------
__global__ void __launch_bounds__(256)
rpab_qkv_proj(const bf16_t* __restrict__ tokB,
              const bf16_t* __restrict__ WT,
              const float* __restrict__ bq,
              const float* __restrict__ bk,
              const float* __restrict__ bv,
              bf16_t* __restrict__ Q,
              bf16_t* __restrict__ K,
              bf16_t* __restrict__ Vt) {
  __shared__ __align__(16) bf16_t Wtile[2][kC * 32];      // 2 x 16 KB

  const int lane  = threadIdx.x & 31;
  const int wave  = threadIdx.x >> 5;
  const int which = blockIdx.y;
  const int b     = blockIdx.z;
  const int rbase = (blockIdx.x * 4 + (wave >> 1)) * 16;
  const int cbase = (wave & 1) * 128;

  const bf16_t* A  = tokB + ((size_t)b * kN + rbase) * kC;
  const bf16_t* Bt = WT + (size_t)which * kC * kC;
  const float* bias = (which == 0) ? bq : (which == 1) ? bk : bv;
  const float scale = (which == 0) ? 0.17677669529663687f : 1.0f;

  v8f acc[8];
  gemm_tdm(A, Bt, Wtile[0], Wtile[1], lane, wave, cbase, acc);

  const int cl = lane & 15;
  const int ro = (lane & 16) ? 8 : 0;
#pragma unroll
  for (int t = 0; t < 8; ++t) {
    const int col = cbase + t * 16 + cl;
    const float bb = bias[col];
#pragma unroll
    for (int i = 0; i < 8; ++i) {
      const int row = rbase + ro + i;
      const float v = (acc[t][i] + bb) * scale;
      if (which == 0)      Q[((size_t)b * kN + row) * kC + col] = (bf16_t)v;
      else if (which == 1) K[((size_t)b * kN + row) * kC + col] = (bf16_t)v;
      else                 Vt[((size_t)b * kC + col) * kN + row] = (bf16_t)v;
    }
  }
}

// ---------------------------------------------------------------------------
// Kernel 4: flash attention, 64-key steps, TDM double-buffered K/V tiles.
// S^T = K_tile x Q^T so exp(S) is already an A-fragment for P x V.
// ---------------------------------------------------------------------------
__global__ void __launch_bounds__(256)
rpab_attention(const bf16_t* __restrict__ Q,
               const bf16_t* __restrict__ K,
               const bf16_t* __restrict__ Vt,
               bf16_t* __restrict__ O) {
  __shared__ __align__(16) bf16_t Ktile[2][64 * 32];      // [key][dh], 2 x 4 KB
  __shared__ __align__(16) bf16_t Vtile[2][32 * 64];      // [dh][key], 2 x 4 KB

  const int b    = blockIdx.z;
  const int head = blockIdx.y;
  const int lane = threadIdx.x & 31;
  const int wave = threadIdx.x >> 5;
  const int qbase = (blockIdx.x * 8 + wave) * 16;

  const bf16_t* Qb  = Q  + (size_t)b * kN * kC;
  const bf16_t* Kb  = K  + (size_t)b * kN * kC + head * kDH;
  const bf16_t* Vtb = Vt + (size_t)b * kC * kN + (size_t)(head * kDH) * kN;

  const v16bf qf = load_frag16x32(Qb + (size_t)qbase * kC + head * kDH, kC, lane);

  const v8f zero = {0.f, 0.f, 0.f, 0.f, 0.f, 0.f, 0.f, 0.f};
  v8f acc0 = zero, acc1 = zero;
  float m = -1e30f, l = 0.0f;

  if (wave == 0) {
    tdm_load_2d(lds_off_of(Ktile[0]), Kb, 32, 64, kC);          // 64 keys x 32 dh
    tdm_load_2d(lds_off_of(Vtile[0]), Vtb, 64, 32, kN);         // 32 dh x 64 keys
  }

  for (int kb = 0; kb < kN; kb += 64) {
    const int cur = (kb >> 6) & 1;
    const bool more = (kb + 64) < kN;
    if (wave == 0) {
      if (more) {
        tdm_load_2d(lds_off_of(Ktile[cur ^ 1]),
                    Kb + (size_t)(kb + 64) * kC, 32, 64, kC);
        tdm_load_2d(lds_off_of(Vtile[cur ^ 1]),
                    Vtb + (kb + 64), 64, 32, kN);
        wait_tensorcnt<2>();                 // current tile done; next in flight
      } else {
        wait_tensorcnt<0>();
      }
    }
    __syncthreads();

    const v16bf kf0  = load_frag16x32(Ktile[cur],               32, lane);
    const v16bf kf1  = load_frag16x32(Ktile[cur] + 16 * 32,     32, lane);
    const v16bf kf2  = load_frag16x32(Ktile[cur] + 32 * 32,     32, lane);
    const v16bf kf3  = load_frag16x32(Ktile[cur] + 48 * 32,     32, lane);
    const v16bf vf00 = load_frag16x32(Vtile[cur],               64, lane);
    const v16bf vf01 = load_frag16x32(Vtile[cur] + 32,          64, lane);
    const v16bf vf10 = load_frag16x32(Vtile[cur] + 16 * 64,     64, lane);
    const v16bf vf11 = load_frag16x32(Vtile[cur] + 16 * 64 + 32, 64, lane);
    __syncthreads();

    v8f t0 = wmma_bf16(kf0, qf, zero);
    v8f t1 = wmma_bf16(kf1, qf, zero);
    v8f t2 = wmma_bf16(kf2, qf, zero);
    v8f t3 = wmma_bf16(kf3, qf, zero);

    float mx = -1e30f;
#pragma unroll
    for (int i = 0; i < 8; ++i)
      mx = fmaxf(mx, fmaxf(fmaxf(t0[i], t1[i]), fmaxf(t2[i], t3[i])));
    mx = fmaxf(mx, __shfl_xor(mx, 16, 32));
    const float mn = fmaxf(m, mx);
    const float f  = __expf(m - mn);
    m = mn;

    v16bf pf0, pf1;
    float ps = 0.0f;
#pragma unroll
    for (int i = 0; i < 8; ++i) {
      const float p0 = __expf(t0[i] - mn);
      const float p1 = __expf(t1[i] - mn);
      const float p2 = __expf(t2[i] - mn);
      const float p3 = __expf(t3[i] - mn);
      ps += (p0 + p1) + (p2 + p3);
      pf0[i] = (bf16_t)p0; pf0[i + 8] = (bf16_t)p1;
      pf1[i] = (bf16_t)p2; pf1[i + 8] = (bf16_t)p3;
    }
    ps += __shfl_xor(ps, 16, 32);
    l = l * f + ps;

#pragma unroll
    for (int i = 0; i < 8; ++i) {
      const int src = (lane & 16) ? (8 + i) : i;
      const float fi = __shfl(f, src, 32);
      acc0[i] *= fi;
      acc1[i] *= fi;
    }

    acc0 = wmma_bf16(pf0, vf00, acc0);
    acc0 = wmma_bf16(pf1, vf01, acc0);
    acc1 = wmma_bf16(pf0, vf10, acc1);
    acc1 = wmma_bf16(pf1, vf11, acc1);
  }

  bf16_t* Ob = O + (size_t)b * kN * kC;
  const int cl = lane & 15;
  const int ro = (lane & 16) ? 8 : 0;
#pragma unroll
  for (int i = 0; i < 8; ++i) {
    const int src = (lane & 16) ? (8 + i) : i;
    const float inv = 1.0f / __shfl(l, src, 32);
    const int row = qbase + ro + i;
    const int col = head * kDH + cl;
    Ob[(size_t)row * kC + col]      = (bf16_t)(acc0[i] * inv);
    Ob[(size_t)row * kC + col + 16] = (bf16_t)(acc1[i] * inv);
  }
}

// ---------------------------------------------------------------------------
// Kernel 5: Z = O x Wo + bo + tok (f32 to workspace), TDM-staged weights.
// ---------------------------------------------------------------------------
__global__ void __launch_bounds__(256)
rpab_out_proj(const bf16_t* __restrict__ O,
              const bf16_t* __restrict__ WoT,
              const float* __restrict__ bo,
              const float* __restrict__ tokF,
              float* __restrict__ Z) {
  __shared__ __align__(16) bf16_t Wtile[2][kC * 32];

  const int lane  = threadIdx.x & 31;
  const int wave  = threadIdx.x >> 5;
  const int b     = blockIdx.y;
  const int rbase = (blockIdx.x * 4 + (wave >> 1)) * 16;
  const int cbase = (wave & 1) * 128;

  const bf16_t* A = O + ((size_t)b * kN + rbase) * kC;
  v8f acc[8];
  gemm_tdm(A, WoT, Wtile[0], Wtile[1], lane, wave, cbase, acc);

  const int cl = lane & 15;
  const int ro = (lane & 16) ? 8 : 0;
#pragma unroll
  for (int t = 0; t < 8; ++t) {
    const int col = cbase + t * 16 + cl;
    const float bb = bo[col];
#pragma unroll
    for (int i = 0; i < 8; ++i) {
      const int row = rbase + ro + i;
      const size_t o = ((size_t)b * kN + row) * kC + col;
      Z[o] = acc[t][i] + bb + tokF[o];
    }
  }
}

// ---------------------------------------------------------------------------
// Kernel 6: row-wise LayerNorm + scatter to dense grid (one row per wave).
// ---------------------------------------------------------------------------
__global__ void __launch_bounds__(256)
rpab_ln_scatter(const float* __restrict__ Z,
                const float* __restrict__ gamma,
                const float* __restrict__ beta,
                const int* __restrict__ idx,
                const int* __restrict__ cnt,
                float* __restrict__ out) {
  const int lane = threadIdx.x & 31;
  const int wave = threadIdx.x >> 5;
  const int b    = blockIdx.y;
  const int row  = blockIdx.x * 8 + wave;
  if (row >= cnt[b]) return;

  const float* z = Z + ((size_t)b * kN + row) * kC + lane * 8;
  float v[8];
  *(float4*)&v[0] = *(const float4*)&z[0];
  *(float4*)&v[4] = *(const float4*)&z[4];

  float s = 0.f, q = 0.f;
#pragma unroll
  for (int j = 0; j < 8; ++j) { s += v[j]; q += v[j] * v[j]; }
#pragma unroll
  for (int off = 1; off < 32; off <<= 1) {
    s += __shfl_xor(s, off, 32);
    q += __shfl_xor(q, off, 32);
  }
  const float mu   = s * (1.0f / kC);
  const float var  = q * (1.0f / kC) - mu * mu;
  const float rstd = rsqrtf(var + 1e-5f);

  const int pos = idx[b * kN + row];
#pragma unroll
  for (int j = 0; j < 8; ++j) {
    const int col = lane * 8 + j;
    const float y = (v[j] - mu) * rstd * gamma[col] + beta[col];
    out[((size_t)b * kC + col) * kHW + pos] = y;
  }
}

// ---------------------------------------------------------------------------
// Host-side launch
// ---------------------------------------------------------------------------
extern "C" void kernel_launch(void* const* d_in, const int* in_sizes, int n_in,
                              void* d_out, int out_size, void* d_ws, size_t ws_size,
                              hipStream_t stream) {
  (void)in_sizes; (void)n_in; (void)ws_size;

  const float* x     = (const float*)d_in[0];
  const float* Wq    = (const float*)d_in[1];
  const float* bq    = (const float*)d_in[2];
  const float* Wk    = (const float*)d_in[3];
  const float* bk    = (const float*)d_in[4];
  const float* Wv    = (const float*)d_in[5];
  const float* bv    = (const float*)d_in[6];
  const float* Wo    = (const float*)d_in[7];
  const float* bo    = (const float*)d_in[8];
  const float* gamma = (const float*)d_in[9];
  const float* beta  = (const float*)d_in[10];
  float* out = (float*)d_out;

  char* ws = (char*)d_ws;
  size_t off = 0;
  auto take = [&](size_t bytes) -> void* {
    void* p = ws + off;
    off = (off + bytes + 255) & ~(size_t)255;
    return p;
  };
  int*    cnt  = (int*)   take((size_t)kB * sizeof(int));
  int*    idx  = (int*)   take((size_t)kB * kN * sizeof(int));
  float*  tokF = (float*) take((size_t)kB * kN * kC * sizeof(float));
  float*  Zf   = (float*) take((size_t)kB * kN * kC * sizeof(float));
  bf16_t* tokB = (bf16_t*)take((size_t)kB * kN * kC * sizeof(bf16_t));
  bf16_t* Qbf  = (bf16_t*)take((size_t)kB * kN * kC * sizeof(bf16_t));
  bf16_t* Kbf  = (bf16_t*)take((size_t)kB * kN * kC * sizeof(bf16_t));
  bf16_t* Vt   = (bf16_t*)take((size_t)kB * kC * kN * sizeof(bf16_t));
  bf16_t* Obf  = (bf16_t*)take((size_t)kB * kN * kC * sizeof(bf16_t));
  bf16_t* WT   = (bf16_t*)take((size_t)4 * kC * kC * sizeof(bf16_t));

  (void)hipMemsetAsync(out, 0, (size_t)out_size * sizeof(float), stream);
  (void)hipMemsetAsync(cnt, 0, (size_t)kB * sizeof(int), stream);

  rpab_prep_weights<<<dim3((kC * kC) / 256, 4), 256, 0, stream>>>(Wq, Wk, Wv, Wo, WT);
  rpab_gather<<<dim3(kHW / 256, kB), 256, 0, stream>>>(x, tokF, tokB, idx, cnt);
  rpab_qkv_proj<<<dim3(kN / 64, 3, kB), 256, 0, stream>>>(
      tokB, WT, bq, bk, bv, Qbf, Kbf, Vt);
  rpab_attention<<<dim3(kN / 128, kNH, kB), 256, 0, stream>>>(Qbf, Kbf, Vt, Obf);
  rpab_out_proj<<<dim3(kN / 64, kB), 256, 0, stream>>>(
      Obf, WT + (size_t)3 * kC * kC, bo, tokF, Zf);
  rpab_ln_scatter<<<dim3(kN / 8, kB), 256, 0, stream>>>(
      Zf, gamma, beta, idx, cnt, out);
}
